// Attention_21457656610959
// MI455X (gfx1250) — matrix operand order repaired and checked by reading, not compiled, not run on hardware
//
#include <hip/hip_runtime.h>
#include <hip/hip_bf16.h>

#define BB 8
#define LL 2048
#define DD 512
#define TDM_KEYS 32

typedef __attribute__((ext_vector_type(16))) __bf16 v16bf;
typedef __attribute__((ext_vector_type(8)))  __bf16 v8bf;
typedef __attribute__((ext_vector_type(8)))  float  v8f;
typedef __attribute__((ext_vector_type(4)))  unsigned int u32x4;
typedef __attribute__((ext_vector_type(4)))  int i32x4;
typedef __attribute__((ext_vector_type(8)))  int i32x8;

#if __has_builtin(__builtin_amdgcn_tensor_load_to_lds)
#define HAVE_TDM 1
#else
#define HAVE_TDM 0
#endif

static __device__ __forceinline__ int laneid() { return threadIdx.x & 31; }

// ---- WMMA fragment loaders (CDNA5 16x16x32 bf16 layouts, ISA 7.12.2) ----

// A (16x32, MxK): lane r=l&15 is row M; lane half g selects K groups.
static __device__ __forceinline__ v16bf a_frag_bf16(const __bf16* src, int ld, int k0) {
  int l = laneid(); int r = l & 15, g = l >> 4;
  const __bf16* p = src + (size_t)r * ld + k0 + g * 8;
  v8bf lo = *(const v8bf*)p;          // K = base+0..7
  v8bf hi = *(const v8bf*)(p + 16);   // K = base+16..23
  return __builtin_shufflevector(lo, hi, 0,1,2,3,4,5,6,7,8,9,10,11,12,13,14,15);
}

static __device__ __forceinline__ v16bf a_frag_f32(const float* src, int ld, int k0) {
  int l = laneid(); int r = l & 15, g = l >> 4;
  const float* p = src + (size_t)r * ld + k0 + g * 8;
  v16bf a;
#pragma unroll
  for (int i = 0; i < 8; ++i) a[i] = (__bf16)p[i];
#pragma unroll
  for (int i = 0; i < 8; ++i) a[8 + i] = (__bf16)p[16 + i];
  return a;
}

// B (32x16, KxN): lane l&15 is column N; lane half g selects K groups (K = g*8+0..7, 16+g*8+0..7).
static __device__ __forceinline__ v16bf b_frag_bf16(const __bf16* src, int ld, int n0) {
  int l = laneid(); int n = l & 15, g = l >> 4;
  const __bf16* p = src + (size_t)(g * 8) * ld + n0 + n;
  v16bf b;
#pragma unroll
  for (int i = 0; i < 8; ++i) b[i] = p[(size_t)i * ld];
#pragma unroll
  for (int i = 0; i < 8; ++i) b[8 + i] = p[(size_t)(16 + i) * ld];
  return b;
}

static __device__ __forceinline__ v8f wmma_bf16(v16bf a, v16bf b, v8f c) {
  return __builtin_amdgcn_wmma_f32_16x16x32_bf16(false, a, false, b, (short)0, c, false, false);
}

#if HAVE_TDM
// TDM: 2D tile load (TDM_KEYS rows x DD bf16 cols) from global to LDS.
// D# group0/group1 packed per CDNA5 ISA 8.3/8.4 (count=1, data_size=2B, type=2).
static __device__ __forceinline__ void tdm_load_vp(const __bf16* g, unsigned lds_byte_addr) {
  unsigned long long ga = (unsigned long long)(size_t)g;
  u32x4 g0 = {};
  g0[0] = 1u;                                   // count=1, no gather
  g0[1] = lds_byte_addr;                        // lds_addr
  g0[2] = (unsigned)ga;                         // global_addr[31:0]
  g0[3] = (unsigned)((ga >> 32) & 0x1FFFFFFull) | (2u << 30);  // addr[56:32] | type=2
  i32x8 g1 = {};
  g1[0] = (int)(1u << 16);                      // data_size = 1 (2 bytes)
  g1[1] = (int)(((unsigned)DD & 0xFFFFu) << 16);          // tensor_dim0 lo
  g1[2] = (int)(((unsigned)LL & 0xFFFFu) << 16);          // dim0 hi | tensor_dim1 lo
  g1[3] = (int)(((unsigned)DD & 0xFFFFu) << 16);          // dim1 hi | tile_dim0 = DD
  g1[4] = (int)TDM_KEYS;                        // tile_dim1 = 32, tile_dim2 = 0
  g1[5] = (int)DD;                              // tensor_dim0_stride low
  i32x4 z4 = {};
#if defined(__clang_major__) && __clang_major__ >= 23
  i32x8 z8 = {};
  __builtin_amdgcn_tensor_load_to_lds(g0, g1, z4, z4, z8, 0);
#else
  __builtin_amdgcn_tensor_load_to_lds(g0, g1, z4, z4, 0);
#endif
}
#endif

// ---- Kernel 0: W[e][d] (f32) -> WT[d][e] (bf16), for q/k/v via blockIdx.y ----
__global__ void wtrans_kernel(const float* __restrict__ Wq, const float* __restrict__ Wk,
                              const float* __restrict__ Wv,
                              __bf16* __restrict__ WqT, __bf16* __restrict__ WkT,
                              __bf16* __restrict__ WvT) {
  const float* src = (blockIdx.y == 0) ? Wq : (blockIdx.y == 1) ? Wk : Wv;
  __bf16* dst = (blockIdx.y == 0) ? WqT : (blockIdx.y == 1) ? WkT : WvT;
  int i = blockIdx.x * blockDim.x + threadIdx.x;   // over D*D
  int e = i / DD, d = i - e * DD;
  dst[(size_t)d * DD + e] = (__bf16)src[(size_t)e * DD + d];
}

// ---- Kernel 1: proj = x @ W^T + b ; output bf16 row-major, or transposed (kpT) ----
__global__ void __launch_bounds__(256)
proj_kernel(const float* __restrict__ x, const __bf16* __restrict__ WT,
            const float* __restrict__ bias, __bf16* __restrict__ dst, int tstore) {
  int wave = threadIdx.x >> 5;
  int wr = wave >> 2, wc = wave & 3;
  int row0 = blockIdx.x * 32 + wr * 16;            // in [0, B*L)
  int col0 = blockIdx.y * 256 + wc * 64;           // in [0, D)
  int l = laneid(); int r = l & 15, g = l >> 4;

  v8f acc[4] = {};
  for (int d0 = 0; d0 < DD; d0 += 32) {
    v16bf a = a_frag_f32(x + (size_t)row0 * DD, DD, d0);
#pragma unroll
    for (int t = 0; t < 4; ++t) {
      v16bf bm = b_frag_bf16(WT + (size_t)d0 * DD, DD, col0 + 16 * t);
      acc[t] = wmma_bf16(a, bm, acc[t]);
    }
  }
#pragma unroll
  for (int t = 0; t < 4; ++t) {
    int col = col0 + 16 * t + r;
    float bv = bias[col];
#pragma unroll
    for (int vr = 0; vr < 8; ++vr) {
      int M = vr + 8 * g;
      int gr = row0 + M;
      float o = acc[t][vr] + bv;
      if (tstore) {  // kpT[b][d][l]
        int bb = gr >> 11, li = gr & (LL - 1);
        dst[(size_t)bb * DD * LL + (size_t)col * LL + li] = (__bf16)o;
      } else {       // row-major [b*L + l][d]
        dst[(size_t)gr * DD + col] = (__bf16)o;
      }
    }
  }
}

// ---- Kernel 2: column softmax stats over q: m[k]=max_q S, s[k]=sum_q exp(S-m) ----
// Wave's entire kpT slice (16 keys x 512 d = 16 fragments) is register-resident.
__global__ void __launch_bounds__(128)
colstats_kernel(const __bf16* __restrict__ qp, const __bf16* __restrict__ kpT,
                float* __restrict__ cmax, float* __restrict__ csum) {
  int b = blockIdx.y;
  int wave = threadIdx.x >> 5;
  int key0 = blockIdx.x * 64 + wave * 16;
  int l = laneid();
  const __bf16* qb = qp + (size_t)b * LL * DD;
  const __bf16* kb = kpT + (size_t)b * DD * LL;

  v16bf kf[16];
#pragma unroll
  for (int i = 0; i < 16; ++i)
    kf[i] = b_frag_bf16(kb + (size_t)(i * 32) * LL, LL, key0);

  float m = -__builtin_inff(), s = 0.f;
  for (int q0 = 0; q0 < LL; q0 += 16) {
    if (q0 + 16 < LL)
      __builtin_prefetch(qb + (size_t)(q0 + 16) * DD + (l << 4), 0, 0);
    v8f c = {};
#pragma unroll
    for (int i = 0; i < 16; ++i) {
      v16bf a = a_frag_bf16(qb + (size_t)q0 * DD, DD, i * 32);
      c = wmma_bf16(a, kf[i], c);
    }
    float tmax = c[0];
#pragma unroll
    for (int i = 1; i < 8; ++i) tmax = fmaxf(tmax, c[i]);
    tmax = fmaxf(tmax, __shfl_xor(tmax, 16, 32));
    float nm = fmaxf(m, tmax);
    float ts = 0.f;
#pragma unroll
    for (int i = 0; i < 8; ++i) ts += __expf(c[i] - nm);
    ts += __shfl_xor(ts, 16, 32);
    s = s * __expf(m - nm) + ts;
    m = nm;
  }
  if (l < 16) {
    cmax[(size_t)b * LL + key0 + l] = m;
    csum[(size_t)b * LL + key0 + l] = s;
  }
}

// ---- Kernel 3: out[q,:] = sum_k exp(S[q,k]-m[k])/den[k] * vp[k,:] ----
// 16 q rows/block, 8 waves split D. S computed cooperatively via ds_add_f32.
// vp tiles staged by the Tensor Data Mover into double-buffered dynamic LDS.
__global__ void __launch_bounds__(256)
out_kernel(const __bf16* __restrict__ qp, const __bf16* __restrict__ kpT,
           const __bf16* __restrict__ vp, const float* __restrict__ cmax,
           const float* __restrict__ csum, float* __restrict__ out) {
  int b = blockIdx.y;
  int q0 = blockIdx.x * 16;
  int wave = threadIdx.x >> 5;          // 0..7
  int dcol0 = wave * 64;
  int l = laneid(); int r = l & 15, g = l >> 4;

  const __bf16* qb = qp + (size_t)b * LL * DD + (size_t)q0 * DD;
  const __bf16* kb = kpT + (size_t)b * DD * LL;
  const __bf16* vb = vp + (size_t)b * LL * DD;

  __shared__ float sS[16 * 32];
  __shared__ float sM[32];
  __shared__ float sR[32];
#if HAVE_TDM
  extern __shared__ __bf16 sVp[];       // 2 buffers of TDM_KEYS x DD
  unsigned sVpBase = (unsigned)(size_t)(&sVp[0]);
#endif

  // qp A-fragments are kc-invariant for this wave's d-range: hoist.
  v16bf aq0 = a_frag_bf16(qb, DD, dcol0);
  v16bf aq1 = a_frag_bf16(qb, DD, dcol0 + 32);

#if HAVE_TDM
  if (wave == 0) {
    tdm_load_vp(vb, sVpBase);
    __builtin_amdgcn_s_wait_tensorcnt(0);
  }
  __syncthreads();
#endif

  v8f acc[4] = {};
  for (int kc = 0; kc < LL; kc += 32) {
#if HAVE_TDM
    int cur = (kc >> 5) & 1;
    if (wave == 0 && (kc + 32) < LL)
      tdm_load_vp(vb + (size_t)(kc + 32) * DD,
                  sVpBase + (unsigned)((cur ^ 1) * TDM_KEYS * DD * 2));
#endif
    if (threadIdx.x < 32) {
      sM[threadIdx.x] = cmax[(size_t)b * LL + kc + threadIdx.x];
      sR[threadIdx.x] = 1.0f / csum[(size_t)b * LL + kc + threadIdx.x];
    }
    for (int i = threadIdx.x; i < 16 * 32; i += 256) sS[i] = 0.f;
    __syncthreads();

    // partial S over this wave's 64-wide d range
    v8f s0 = {}, s1 = {};
    {
      v16bf b0 = b_frag_bf16(kb + (size_t)dcol0 * LL, LL, kc);
      v16bf b1 = b_frag_bf16(kb + (size_t)dcol0 * LL, LL, kc + 16);
      s0 = wmma_bf16(aq0, b0, s0);
      s1 = wmma_bf16(aq0, b1, s1);
      b0 = b_frag_bf16(kb + (size_t)(dcol0 + 32) * LL, LL, kc);
      b1 = b_frag_bf16(kb + (size_t)(dcol0 + 32) * LL, LL, kc + 16);
      s0 = wmma_bf16(aq1, b0, s0);
      s1 = wmma_bf16(aq1, b1, s1);
    }
#pragma unroll
    for (int vr = 0; vr < 8; ++vr) {
      int M = vr + 8 * g;
      atomicAdd(&sS[M * 32 + r], s0[vr]);
      atomicAdd(&sS[M * 32 + 16 + r], s1[vr]);
    }
    __syncthreads();

    // rebuild P (16x32) A-fragment from LDS with column softmax applied
    v16bf pf;
#pragma unroll
    for (int h = 0; h < 16; ++h) {
      int K = (h < 8) ? (g * 8 + h) : (16 + g * 8 + (h - 8));
      float sv = sS[r * 32 + K];
      pf[h] = (__bf16)(__expf(sv - sM[K]) * sR[K]);
    }
    __syncthreads();

#if HAVE_TDM
    const __bf16* vsrc = &sVp[cur * TDM_KEYS * DD];
#else
    const __bf16* vsrc = vb + (size_t)kc * DD;
#endif
#pragma unroll
    for (int t = 0; t < 4; ++t) {
      v16bf bm = b_frag_bf16(vsrc, DD, dcol0 + 16 * t);
      acc[t] = wmma_bf16(pf, bm, acc[t]);
    }

#if HAVE_TDM
    if (wave == 0) __builtin_amdgcn_s_wait_tensorcnt(0);
    __syncthreads();
#endif
  }

  float* ob = out + (size_t)b * LL * DD;
#pragma unroll
  for (int t = 0; t < 4; ++t)
#pragma unroll
    for (int vr = 0; vr < 8; ++vr) {
      int M = vr + 8 * g;
      ob[(size_t)(q0 + M) * DD + dcol0 + 16 * t + r] = acc[t][vr];
    }
}

extern "C" void kernel_launch(void* const* d_in, const int* in_sizes, int n_in,
                              void* d_out, int out_size, void* d_ws, size_t ws_size,
                              hipStream_t stream) {
  (void)in_sizes; (void)n_in; (void)out_size; (void)ws_size;
  const float* q  = (const float*)d_in[0];
  const float* k  = (const float*)d_in[1];
  const float* v  = (const float*)d_in[2];
  const float* Wq = (const float*)d_in[3];
  const float* bq = (const float*)d_in[4];
  const float* Wk = (const float*)d_in[5];
  const float* bk = (const float*)d_in[6];
  const float* Wv = (const float*)d_in[7];
  const float* bv = (const float*)d_in[8];
  float* out = (float*)d_out;

  char* ws = (char*)d_ws;
  size_t off = 0;
  __bf16* WqT = (__bf16*)(ws + off); off += (size_t)DD * DD * 2;
  __bf16* WkT = (__bf16*)(ws + off); off += (size_t)DD * DD * 2;
  __bf16* WvT = (__bf16*)(ws + off); off += (size_t)DD * DD * 2;
  __bf16* qp  = (__bf16*)(ws + off); off += (size_t)BB * LL * DD * 2;
  __bf16* kpT = (__bf16*)(ws + off); off += (size_t)BB * LL * DD * 2;
  __bf16* vp  = (__bf16*)(ws + off); off += (size_t)BB * LL * DD * 2;
  float* cmax = (float*)(ws + off);  off += (size_t)BB * LL * 4;
  float* csum = (float*)(ws + off);  off += (size_t)BB * LL * 4;

  wtrans_kernel<<<dim3((DD * DD) / 256, 3), 256, 0, stream>>>(Wq, Wk, Wv, WqT, WkT, WvT);

  dim3 pg((BB * LL) / 32, DD / 256);
  proj_kernel<<<pg, 256, 0, stream>>>(q, WqT, bq, qp, 0);
  proj_kernel<<<pg, 256, 0, stream>>>(k, WkT, bk, kpT, 1);
  proj_kernel<<<pg, 256, 0, stream>>>(v, WvT, bv, vp, 0);

  colstats_kernel<<<dim3(LL / 64, BB), 128, 0, stream>>>(qp, kpT, cmax, csum);

  size_t dynLds = (size_t)2 * TDM_KEYS * DD * 2;   // double-buffered vp tiles
  out_kernel<<<dim3(LL / 16, BB), 256, dynLds, stream>>>(qp, kpT, vp, cmax, csum, out);
}
